// model2_33097017983662
// MI455X (gfx1250) — matrix operand order repaired and checked by reading, not compiled, not run on hardware
//
#include <hip/hip_runtime.h>

typedef __attribute__((ext_vector_type(16))) _Float16 v16h;
typedef __attribute__((ext_vector_type(8)))  float    v8f;

#define KK        16
#define K2        256
#define DD        88
#define DP        96
#define BT        16
#define NTHREADS  256

// A-matrix 16-bit 16x32 per-lane K index (ISA 7.12.2): lane m holds K (e&7)+(hi?8:0)+(e>=8?16:0)
__device__ __forceinline__ int kA(int e, int hi) {
  return (e & 7) + (hi ? 8 : 0) + ((e >> 3) << 4);
}
// B-matrix 16-bit 32x16 per-lane K index: same as A, lane n holds K (e&7)+(hi?8:0)+(e>=8?16:0)
__device__ __forceinline__ int kB(int e, int hi) {
  return (e & 7) + (hi ? 8 : 0) + ((e >> 3) << 4);
}
__device__ __forceinline__ v8f wmma16(v16h a, v16h b, v8f c) {
  return __builtin_amdgcn_wmma_f32_16x16x32_f16(false, a, false, b, (short)0, c,
                                                false, false);
}

__global__ void fhmm_zero(float* out) { out[0] = 0.f; }

__global__ void __launch_bounds__(NTHREADS, 1)
fhmm_forward(const float* __restrict__ seq,
             const float* __restrict__ pw,
             const float* __restrict__ px,
             const float* __restrict__ wi,
             const float* __restrict__ xi,
             const float* __restrict__ py,
             const int*   __restrict__ lengths,
             const int*   __restrict__ mb,
             const unsigned char* __restrict__ mask,
             float* __restrict__ out,
             int T)
{
  __shared__ _Float16 sSeq[BT * DP];     // 3 KB   seq tile (f16, D padded to 96)
  __shared__ float    sLogB[BT * K2];    // 16 KB  emission log-probs for tile
  __shared__ float    sAlpha[BT * K2];   // 16 KB  normalized alpha (prob domain)
  __shared__ float    sAP[BT * K2];      // 16 KB  step1 temp / alpha-pre (aliased)
  __shared__ float    sLps[K2];          // 1 KB   sum_d log1p(-p)
  __shared__ float    sRed[BT * 16];     // 1 KB   reduction scratch
  __shared__ float    sEll[BT];          // running log-scale per sequence
  __shared__ int      sLen[BT];
  __shared__ int      sRow[BT];
  __shared__ int      sMaxLen;

  const int tid  = threadIdx.x;
  const int lane = tid & 31;             // wave32
  const int wv   = tid >> 5;             // 8 waves
  const int hi   = (lane >> 4) & 1;
  const int nn   = lane & 15;
  const int bq   = blockIdx.x * BT;

  // ---------------- one-time setup ----------------
  for (int k2i = tid; k2i < K2; k2i += NTHREADS) {
    float s = 0.f;
    for (int d = 0; d < DD; ++d) s += log1pf(-py[k2i * DD + d]);
    sLps[k2i] = s;
  }
  if (tid < BT) {
    int r = mb[bq + tid];
    sRow[tid] = r;
    sLen[tid] = lengths[r];
    sEll[tid] = 0.f;
  }
  __syncthreads();
  if (tid == 0) {
    int m = 0;
    for (int b = 0; b < BT; ++b) m = (sLen[b] > m) ? sLen[b] : m;
    sMaxLen = m;
  }
  __syncthreads();
  const int tMax = sMaxLen;

  // Hoist emission B-operands (logit(p), f16) into registers for the whole T loop.
  // Fully unrolled so bP stays in VGPRs (dynamic indexing would force scratch spills).
  v16h bP[2][3];
  #pragma unroll
  for (int ti = 0; ti < 2; ++ti) {
    int k2i = (wv * 2 + ti) * 16 + nn;
    #pragma unroll
    for (int kc = 0; kc < 3; ++kc) {
      v16h v = {};
      #pragma unroll
      for (int e = 0; e < 16; ++e) {
        int d = kc * 32 + kB(e, hi);
        float val = 0.f;
        if (d < DD) {
          float p = py[k2i * DD + d];
          val = __logf(p) - log1pf(-p);
        }
        v[e] = (_Float16)val;
      }
      bP[ti][kc] = v;
    }
  }
  // Hoist transition B-operands (prob domain), K padded 16 -> 32 with zeros.
  v16h bX = {}, bW = {};
  #pragma unroll
  for (int e = 0; e < 16; ++e) {
    int k = kB(e, hi);
    if (k < KK) {
      bX[e] = (_Float16)px[k * KK + nn];   // B[k=x][n=x'] = X[x][x']
      bW[e] = (_Float16)pw[k * KK + nn];   // B[k=w][n=w'] = W[w][w']
    }
  }

  const int b_el = tid >> 4;               // elementwise phase: 16 threads / sequence
  const int j_el = tid & 15;

  for (int t = 0; t < T; ++t) {
    if (t > 0 && t >= tMax) break;         // all sequences in tile finished

    // ---- stage seq[mb[b], t, :] tile into LDS as f16 ----
    for (int i = tid; i < BT * DP; i += NTHREADS) {
      int b = i / DP, d = i - b * DP;
      float v = 0.f;
      if (d < DD) {
        const float* p = &seq[((size_t)sRow[b] * T + t) * DD + d];
        v = *p;
        if (t + 1 < T) __builtin_prefetch(p + DD, 0, 1);
      }
      sSeq[b * DP + d] = (_Float16)v;
    }
    __syncthreads();

    // ---- emission GEMM: [16 x 96] x [96 x 256] -> sLogB (f32 accum) ----
    v16h aS[3];
    #pragma unroll
    for (int kc = 0; kc < 3; ++kc) {
      v16h v = {};
      #pragma unroll
      for (int e = 0; e < 16; ++e)
        v[e] = sSeq[nn * DP + kc * 32 + kA(e, hi)];
      aS[kc] = v;
    }
    #pragma unroll
    for (int ti = 0; ti < 2; ++ti) {
      v8f c = {};
      c = wmma16(aS[0], bP[ti][0], c);
      c = wmma16(aS[1], bP[ti][1], c);
      c = wmma16(aS[2], bP[ti][2], c);
      int nt = wv * 2 + ti;
      #pragma unroll
      for (int i = 0; i < 8; ++i) {
        int m   = i + hi * 8;
        int k2i = nt * 16 + nn;
        sLogB[m * K2 + k2i] = c[i] + sLps[k2i];
      }
    }
    __syncthreads();

    if (t == 0) {
      // alpha_pre = outer(w_init, x_init)
      for (int i = tid; i < BT * K2; i += NTHREADS) {
        int k2i = i & (K2 - 1);
        sAP[i] = wi[k2i >> 4] * xi[k2i & 15];
      }
      __syncthreads();
    } else {
      // ---- factorized transition: two 16x16 matmuls per sequence ----
      // step 1: t1[b][w][x'] = sum_x alpha[b][w][x] * X[x][x']
      #pragma unroll
      for (int bi = 0; bi < 2; ++bi) {
        int b = wv * 2 + bi;
        v16h a = {};
        #pragma unroll
        for (int e = 0; e < 16; ++e) {
          int k = kA(e, hi);
          if (k < KK) a[e] = (_Float16)sAlpha[b * K2 + nn * 16 + k]; // A[m=w][k=x]
        }
        v8f c = {};
        c = wmma16(a, bX, c);
        #pragma unroll
        for (int i = 0; i < 8; ++i)
          sAP[b * K2 + (i + hi * 8) * 16 + nn] = c[i];   // [w=m][x'=n]
      }
      __syncthreads();
      // step 2: ap[b][w'][x'] = sum_w W[w][w'] * t1[b][w][x']   (in-place over sAP)
      #pragma unroll
      for (int bi = 0; bi < 2; ++bi) {
        int b = wv * 2 + bi;
        v16h a = {};
        #pragma unroll
        for (int e = 0; e < 16; ++e) {
          int k = kA(e, hi);
          if (k < KK) a[e] = (_Float16)sAP[b * K2 + k * 16 + nn]; // A[m=x'][k=w]
        }
        v8f c = {};
        c = wmma16(a, bW, c);
        #pragma unroll
        for (int i = 0; i < 8; ++i)
          sAP[b * K2 + nn * 16 + (i + hi * 8)] = c[i];  // [w'=n][x'=m]
      }
      __syncthreads();
    }

    // ---- emission weighting + rescaling (scaled forward algorithm) ----
    float mx = -3.0e38f;
    #pragma unroll
    for (int s = 0; s < 16; ++s)
      mx = fmaxf(mx, sLogB[b_el * K2 + s * 16 + j_el]);
    sRed[b_el * 16 + j_el] = mx;
    __syncthreads();
    float m_b = sRed[b_el * 16];
    #pragma unroll
    for (int s = 1; s < 16; ++s) m_b = fmaxf(m_b, sRed[b_el * 16 + s]);

    float vals[16];
    float loc = 0.f;
    #pragma unroll
    for (int s = 0; s < 16; ++s) {
      int k2i = s * 16 + j_el;
      float v = sAP[b_el * K2 + k2i] * __expf(sLogB[b_el * K2 + k2i] - m_b);
      vals[s] = v;
      loc += v;
    }
    __syncthreads();                       // sRed reuse
    sRed[b_el * 16 + j_el] = loc;
    __syncthreads();
    float c_b = 0.f;
    #pragma unroll
    for (int s = 0; s < 16; ++s) c_b += sRed[b_el * 16 + s];

    if (t < sLen[b_el]) {                  // length mask: frozen past end
      float inv = 1.f / c_b;
      #pragma unroll
      for (int s = 0; s < 16; ++s)
        sAlpha[b_el * K2 + s * 16 + j_el] = vals[s] * inv;
      if (j_el == 0) sEll[b_el] += m_b + __logf(c_b);
    }
    __syncthreads();
  }

  // alpha is normalized, so loglik = accumulated log-scale
  if (tid < BT) {
    if (mask[bq + tid]) atomicAdd(out, sEll[tid]);
  }
}

extern "C" void kernel_launch(void* const* d_in, const int* in_sizes, int n_in,
                              void* d_out, int out_size, void* d_ws, size_t ws_size,
                              hipStream_t stream) {
  const float* seq = (const float*)d_in[0];
  const float* pw  = (const float*)d_in[1];
  const float* px  = (const float*)d_in[2];
  const float* wi  = (const float*)d_in[3];
  const float* xi  = (const float*)d_in[4];
  const float* py  = (const float*)d_in[5];
  const int* lengths = (const int*)d_in[6];
  const int* mb      = (const int*)d_in[7];
  const unsigned char* mask = (const unsigned char*)d_in[8];
  float* out = (float*)d_out;

  long long N = in_sizes[6];
  long long B = in_sizes[7];
  int T = (int)((long long)in_sizes[0] / (N * DD));

  hipLaunchKernelGGL(fhmm_zero, dim3(1), dim3(1), 0, stream, out);
  hipLaunchKernelGGL(fhmm_forward, dim3((unsigned)(B / BT)), dim3(NTHREADS), 0,
                     stream, seq, pw, px, wi, xi, py, lengths, mb, mask, out, T);
}
